// SimilarityFunctionWithTorch_79852031967548
// MI455X (gfx1250) — matrix operand order repaired and checked by reading, not compiled, not run on hardware
//
#include <hip/hip_runtime.h>

typedef __attribute__((ext_vector_type(16))) __bf16 v16bf;
typedef __attribute__((ext_vector_type(8)))  __bf16 v8bf;
typedef __attribute__((ext_vector_type(8)))  float  v8f;

#define MDIM 8192
#define NDIM 8192
#define KDIM 256

#define WMMA_BF16(A, B, C) \
    __builtin_amdgcn_wmma_f32_16x16x32_bf16(false, (A), false, (B), (short)0, (C), false, false)

// ---------------------------------------------------------------------------
// Kernel 1: row-normalize both matrices and split each f32 into bf16 hi/lo.
// rows 0..8191 -> refer -> A planes ; rows 8192..16383 -> compr -> B planes.
// One wave32 per row of 256 floats (8 per lane).
// ---------------------------------------------------------------------------
__global__ __launch_bounds__(256) void normalize_split_kernel(
    const float* __restrict__ refer, const float* __restrict__ compr,
    __bf16* __restrict__ a_hi, __bf16* __restrict__ a_lo,
    __bf16* __restrict__ b_hi, __bf16* __restrict__ b_lo)
{
    const int wave = threadIdx.x >> 5;
    const int lane = threadIdx.x & 31;
    const int row  = blockIdx.x * 8 + wave;   // 0 .. 16383

    const float* src;
    __bf16 *dhi, *dlo;
    int r;
    if (row < MDIM) { src = refer; dhi = a_hi; dlo = a_lo; r = row; }
    else            { src = compr; dhi = b_hi; dlo = b_lo; r = row - MDIM; }

    const float* p = src + (size_t)r * KDIM;

    float x[8];
    float ss = 0.0f;
#pragma unroll
    for (int j = 0; j < 8; ++j) {
        x[j] = p[lane + 32 * j];
        ss += x[j] * x[j];
    }
    // wave32 reduction
#pragma unroll
    for (int off = 16; off > 0; off >>= 1)
        ss += __shfl_xor(ss, off, 32);

    const float scale = 1.0f / fmaxf(sqrtf(ss), 1e-8f);

    __bf16* oh = dhi + (size_t)r * KDIM;
    __bf16* ol = dlo + (size_t)r * KDIM;
#pragma unroll
    for (int j = 0; j < 8; ++j) {
        float v  = x[j] * scale;
        __bf16 h = (__bf16)v;            // RNE
        __bf16 l = (__bf16)(v - (float)h);
        oh[lane + 32 * j] = h;
        ol[lane + 32 * j] = l;
    }
}

// ---------------------------------------------------------------------------
// Kernel 2: bf16x3 WMMA GEMM. 128x128 tile per block, 8 waves (4 M x 2 N),
// 32x64 per wave = 2x4 grid of 16x16 accumulators. K loop: 8 steps of 32.
// D = Ahi*Bhi + Ahi*Blo + Alo*Bhi  (f32 accumulate).
// ---------------------------------------------------------------------------
__global__ __launch_bounds__(256) void cosine_gemm_kernel(
    const __bf16* __restrict__ a_hi, const __bf16* __restrict__ a_lo,
    const __bf16* __restrict__ b_hi, const __bf16* __restrict__ b_lo,
    float* __restrict__ out)
{
    const int lane    = threadIdx.x & 31;
    const int wave    = threadIdx.x >> 5;
    const int waveRow = wave & 3;   // 4 waves along M, 32 rows each
    const int waveCol = wave >> 2;  // 2 waves along N, 64 cols each

    const int blockM = blockIdx.y * 128;
    const int blockN = blockIdx.x * 128;
    const int laneRC = lane & 15;       // row (A) / col (B) within 16-tile
    const int laneHi = lane >> 4;       // 0 or 1: K half selector

    // Operand layouts used here (contiguous k order, same for A and B):
    //  A (16x32 bf16): lane group 0 holds K 0..15 contiguous, group 1 K 16..31
    //  B (32x16 bf16): lane group 0 holds K 0..15 contiguous, group 1 K 16..31
    const int aK = laneHi << 4;   // first 8-elem chunk offset; second at +8
    const int bK = laneHi << 4;   // contiguous 16-elem chunk offset

    const __bf16* pah[2]; const __bf16* pal[2];
    const __bf16* pbh[4]; const __bf16* pbl[4];
#pragma unroll
    for (int mi = 0; mi < 2; ++mi) {
        const size_t rowA = (size_t)(blockM + waveRow * 32 + mi * 16 + laneRC) * KDIM;
        pah[mi] = a_hi + rowA + aK;
        pal[mi] = a_lo + rowA + aK;
    }
#pragma unroll
    for (int ni = 0; ni < 4; ++ni) {
        const size_t rowB = (size_t)(blockN + waveCol * 64 + ni * 16 + laneRC) * KDIM;
        pbh[ni] = b_hi + rowB + bK;
        pbl[ni] = b_lo + rowB + bK;
    }

    v8f acc[2][4];
#pragma unroll
    for (int mi = 0; mi < 2; ++mi)
#pragma unroll
        for (int ni = 0; ni < 4; ++ni)
            acc[mi][ni] = (v8f){0.f, 0.f, 0.f, 0.f, 0.f, 0.f, 0.f, 0.f};

    for (int k0 = 0; k0 < KDIM; k0 += 32) {
        v16bf Ah[2], Al[2], Bh[4], Bl[4];
#pragma unroll
        for (int mi = 0; mi < 2; ++mi) {
            v8bf h0 = *(const v8bf*)(pah[mi] + k0);
            v8bf h1 = *(const v8bf*)(pah[mi] + k0 + 8);
            Ah[mi] = __builtin_shufflevector(h0, h1, 0,1,2,3,4,5,6,7,8,9,10,11,12,13,14,15);
            v8bf l0 = *(const v8bf*)(pal[mi] + k0);
            v8bf l1 = *(const v8bf*)(pal[mi] + k0 + 8);
            Al[mi] = __builtin_shufflevector(l0, l1, 0,1,2,3,4,5,6,7,8,9,10,11,12,13,14,15);
        }
#pragma unroll
        for (int ni = 0; ni < 4; ++ni) {
            Bh[ni] = *(const v16bf*)(pbh[ni] + k0);
            Bl[ni] = *(const v16bf*)(pbl[ni] + k0);
        }

        // Prefetch the next K slab (operands are L2-resident after first use;
        // this mainly primes the WGP-level cache).
        if (k0 + 32 < KDIM) {
#pragma unroll
            for (int mi = 0; mi < 2; ++mi)
                __builtin_prefetch(pah[mi] + k0 + 32, 0, 1);
#pragma unroll
            for (int ni = 0; ni < 4; ++ni)
                __builtin_prefetch(pbh[ni] + k0 + 32, 0, 1);
        }

#pragma unroll
        for (int mi = 0; mi < 2; ++mi) {
#pragma unroll
            for (int ni = 0; ni < 4; ++ni) {
                acc[mi][ni] = WMMA_BF16(Ah[mi], Bh[ni], acc[mi][ni]);
                acc[mi][ni] = WMMA_BF16(Ah[mi], Bl[ni], acc[mi][ni]);
                acc[mi][ni] = WMMA_BF16(Al[mi], Bh[ni], acc[mi][ni]);
            }
        }
    }

    // C/D layout: VGPR r -> row (lane>>4)*8 + r, col lane&15.
#pragma unroll
    for (int mi = 0; mi < 2; ++mi) {
#pragma unroll
        for (int ni = 0; ni < 4; ++ni) {
            const int rowBase = blockM + waveRow * 32 + mi * 16 + laneHi * 8;
            const int col     = blockN + waveCol * 64 + ni * 16 + laneRC;
            float* o = out + (size_t)rowBase * NDIM + col;
#pragma unroll
            for (int r = 0; r < 8; ++r)
                o[(size_t)r * NDIM] = acc[mi][ni][r];
        }
    }
}

// ---------------------------------------------------------------------------
extern "C" void kernel_launch(void* const* d_in, const int* in_sizes, int n_in,
                              void* d_out, int out_size, void* d_ws, size_t ws_size,
                              hipStream_t stream)
{
    (void)in_sizes; (void)n_in; (void)out_size; (void)ws_size;

    const float* compr = (const float*)d_in[0];  // [8192, 256] -> B (N dim)
    const float* refer = (const float*)d_in[1];  // [8192, 256] -> A (M dim)
    float* out = (float*)d_out;                  // [M, N] f32

    // Workspace: 4 bf16 planes of 8192*256 = 4 MB each -> 16 MB total.
    __bf16* a_hi = (__bf16*)d_ws;
    __bf16* a_lo = a_hi + (size_t)MDIM * KDIM;
    __bf16* b_hi = a_lo + (size_t)MDIM * KDIM;
    __bf16* b_lo = b_hi + (size_t)NDIM * KDIM;

    // 16384 rows, 8 waves (rows) per 256-thread block.
    normalize_split_kernel<<<(2 * MDIM) / 8, 256, 0, stream>>>(
        refer, compr, a_hi, a_lo, b_hi, b_lo);

    dim3 grid(NDIM / 128, MDIM / 128);  // 64 x 64 blocks
    cosine_gemm_kernel<<<grid, 256, 0, stream>>>(a_hi, a_lo, b_hi, b_lo, out);
}
